// MultiHeadAttention_30502857736278
// MI455X (gfx1250) — compile-verified, matrix operand order checked by reading, not run on hardware
//
#include <hip/hip_runtime.h>

// Problem constants (from reference)
#define Bc  2
#define Tc  2048
#define Dc  1024
#define Hc  16
#define DKc 64

typedef __attribute__((ext_vector_type(16))) __bf16 v16bf;
typedef __attribute__((ext_vector_type(8)))  float  v8f;

__device__ __forceinline__ unsigned short f2bfu(float f) {
  union { float f; unsigned u; } x; x.f = f;
  unsigned r = x.u + 0x7FFFu + ((x.u >> 16) & 1u);   // round-to-nearest-even
  return (unsigned short)(r >> 16);
}
__device__ __forceinline__ __bf16 f2bf(float f) {
  union { unsigned short s; __bf16 b; } y; y.s = f2bfu(f);
  return y.b;
}

// --- gfx1250 async global->LDS copy (ASYNCcnt-tracked, per ISA ch.10/§15.18)
// lds_off = group-segment byte offset (low 32 bits of the flat shared addr;
// hardware adds the wave's LDS_BASE itself).
__device__ __forceinline__ void async_ld_b128(unsigned lds_off, const void* g) {
  asm volatile("global_load_async_to_lds_b128 %0, %1, off"
               :: "v"(lds_off), "v"(g) : "memory");
}
__device__ __forceinline__ void wait_async0() {
  asm volatile("s_wait_asynccnt 0x0" ::: "memory");
}

// 16x32 bf16 operand fragment per CDNA5 ISA layout (used for A, and for B
// via strides: element(row, k) = p[row*row_stride + k*k_stride]).
// lanes 0-15:  row = row_base+lane,    K = {k_base+0..7, k_base+16..23}
// lanes 16-31: row = row_base+lane-16, K = {k_base+8..15, k_base+24..31}
__device__ __forceinline__ v16bf frag16x32(const __bf16* p, int row_stride,
                                           int k_stride, int row_base, int k_base) {
  const int lane = threadIdx.x & 31;
  const int r  = row_base + (lane & 15);
  const int k0 = k_base + ((lane & 16) ? 8 : 0);
  const __bf16* q = p + r * row_stride;
  v16bf f;
#pragma unroll
  for (int i = 0; i < 8; ++i) f[i] = q[(k0 + i) * k_stride];
#pragma unroll
  for (int i = 0; i < 8; ++i) f[8 + i] = q[(k0 + 16 + i) * k_stride];
  return f;
}

__device__ __forceinline__ v8f wmma_bf16(v16bf a, v16bf b, v8f c) {
  return __builtin_amdgcn_wmma_f32_16x16x32_bf16(false, a, false, b,
                                                 (short)0, c, false, false);
}

// ---------------------------------------------------------------------------
// Y = X @ W^T.   X:[M,K] f32, W:[N,K] f32.
// mode 0: write bf16 head-split [B, H, T, DK]   (for Q/K/V projections)
// mode 1: write f32 row-major [M, N]            (for the output projection)
// Block: 256 threads = 8 waves, tile 128(M) x 128(N), K-step 32.
// Waves: 4 M-strips x 2 N-strips; each wave owns 32x64 (2x4 WMMA tiles).
// ---------------------------------------------------------------------------
__global__ __launch_bounds__(256)
void gemm_xwt_wmma(const float* __restrict__ X, const float* __restrict__ W,
                   unsigned short* __restrict__ Ybf, float* __restrict__ Yf,
                   int M, int N, int K, int mode) {
  __shared__ __align__(16) __bf16 sA[128 * 32];
  __shared__ __align__(16) __bf16 sB[128 * 32];
  const int tid = threadIdx.x;
  const int m0 = blockIdx.y * 128;
  const int n0 = blockIdx.x * 128;
  const int wave = tid >> 5;
  const int wm = wave & 3;   // 4 M-strips of 32
  const int wn = wave >> 2;  // 2 N-strips of 64

  v8f zero = {};
  v8f acc[2][4];
#pragma unroll
  for (int i = 0; i < 2; ++i)
#pragma unroll
    for (int j = 0; j < 4; ++j) acc[i][j] = zero;

  for (int k0 = 0; k0 < K; k0 += 32) {
    // A tile: 128x32 f32 -> bf16 LDS (1024 float4 loads, 4/thread)
    for (int i = tid; i < 1024; i += 256) {
      int r = i >> 3, c4 = (i & 7) << 2;
      float4 f = *(const float4*)(X + (size_t)(m0 + r) * K + k0 + c4);
      __bf16* dst = sA + r * 32 + c4;
      dst[0] = f2bf(f.x); dst[1] = f2bf(f.y); dst[2] = f2bf(f.z); dst[3] = f2bf(f.w);
    }
    // B tile (rows of W): 128x32 f32 -> bf16 LDS (1024 float4 loads, 4/thread)
    for (int i = tid; i < 1024; i += 256) {
      int r = i >> 3, c4 = (i & 7) << 2;
      float4 f = *(const float4*)(W + (size_t)(n0 + r) * K + k0 + c4);
      __bf16* dst = sB + r * 32 + c4;
      dst[0] = f2bf(f.x); dst[1] = f2bf(f.y); dst[2] = f2bf(f.z); dst[3] = f2bf(f.w);
    }
    __syncthreads();

    v16bf a0 = frag16x32(sA, 32, 1, wm * 32, 0);
    v16bf a1 = frag16x32(sA, 32, 1, wm * 32 + 16, 0);
#pragma unroll
    for (int j = 0; j < 4; ++j) {
      v16bf bj = frag16x32(sB, 32, 1, wn * 64 + j * 16, 0);
      acc[0][j] = wmma_bf16(a0, bj, acc[0][j]);
      acc[1][j] = wmma_bf16(a1, bj, acc[1][j]);
    }
    __syncthreads();
  }

  const int lane = tid & 31;
  const int colh = lane & 15;
  const int rbase = (lane & 16) ? 8 : 0;
#pragma unroll
  for (int ti = 0; ti < 2; ++ti)
#pragma unroll
    for (int tj = 0; tj < 4; ++tj)
#pragma unroll
      for (int r = 0; r < 8; ++r) {
        int m = m0 + wm * 32 + ti * 16 + rbase + r;
        int n = n0 + wn * 64 + tj * 16 + colh;
        float val = acc[ti][tj][r];
        if (mode == 0) {
          int b = m / Tc, t = m - b * Tc;
          int h = n >> 6, dk = n & 63;
          Ybf[(((size_t)b * Hc + h) * Tc + t) * DKc + dk] = f2bfu(val);
        } else {
          Yf[(size_t)m * N + n] = val;
        }
      }
}

// ---------------------------------------------------------------------------
// Flash attention with online softmax.
// Grid: (T/64, H, B). Block: 128 threads = 4 waves, each wave owns a 16-row
// query strip of a 64-query tile. Q/K/V are bf16 head-split [B,H,T,64].
// Tiles stream through LDS via global_load_async_to_lds_b128 (ASYNCcnt).
// Bias [H,T,T] streamed once (prefetched one tile ahead); causal + key-pad
// masks applied analytically.
// ---------------------------------------------------------------------------
__global__ __launch_bounds__(128)
void flash_attn_wmma(const unsigned short* __restrict__ Qp,
                     const unsigned short* __restrict__ Kp,
                     const unsigned short* __restrict__ Vp,
                     const float* __restrict__ bias,
                     const unsigned char* __restrict__ pad,
                     float* __restrict__ O) {
  __shared__ __align__(16) __bf16 sQ[64 * 64];
  __shared__ __align__(16) __bf16 sK[64 * 64];
  __shared__ __align__(16) __bf16 sV[64 * 64];
  __shared__ __align__(16) __bf16 sP[4][16 * 64];
  __shared__ unsigned char spad[64];

  const int tid  = threadIdx.x;
  const int lane = tid & 31;
  const int wq   = tid >> 5;      // wave: 16-row query strip
  const int qt   = blockIdx.x;    // 64-query tile index
  const int h    = blockIdx.y;
  const int b    = blockIdx.z;
  const int q0   = qt * 64;

  const __bf16* Qbh = (const __bf16*)Qp + ((size_t)b * Hc + h) * Tc * DKc;
  const __bf16* Kbh = (const __bf16*)Kp + ((size_t)b * Hc + h) * Tc * DKc;
  const __bf16* Vbh = (const __bf16*)Vp + ((size_t)b * Hc + h) * Tc * DKc;
  const float*  biash = bias + (size_t)h * Tc * Tc;

  const unsigned sQo = (unsigned)(uintptr_t)sQ;
  const unsigned sKo = (unsigned)(uintptr_t)sK;
  const unsigned sVo = (unsigned)(uintptr_t)sV;

  // Q tile: contiguous 64x64 bf16 (8 KB) -> 4 async B128 copies per thread
  for (int i = tid; i < 512; i += 128)
    async_ld_b128(sQo + i * 16, (const char*)(Qbh + (size_t)q0 * DKc) + i * 16);

  const int colh  = lane & 15;
  const int rbase = (lane & 16) ? 8 : 0;

  float mrow[8], lrow[8];
#pragma unroll
  for (int i = 0; i < 8; ++i) { mrow[i] = -3.0e38f; lrow[i] = 0.0f; }
  v8f zero = {};
  v8f o[4];
  o[0] = zero; o[1] = zero; o[2] = zero; o[3] = zero;

  for (int kt = 0; kt <= qt; ++kt) {       // causal: skip tiles fully masked
    const int k0 = kt * 64;
    __syncthreads();                        // sK/sV/spad reuse from prev iter
    for (int i = tid; i < 512; i += 128) {
      async_ld_b128(sKo + i * 16, (const char*)(Kbh + (size_t)k0 * DKc) + i * 16);
      async_ld_b128(sVo + i * 16, (const char*)(Vbh + (size_t)k0 * DKc) + i * 16);
    }
    if (tid < 64) spad[tid] = pad[(size_t)b * Tc + k0 + tid];
    // prefetch next key tile's bias block (64 rows x 64 cols f32)
    if (kt < qt) {
      const int pr = tid & 63, pc = (tid >> 6) << 5;
      __builtin_prefetch(&biash[(size_t)(q0 + pr) * Tc + k0 + 64 + pc], 0, 1);
    }
    wait_async0();                          // my async tile writes done
    __syncthreads();                        // everyone's writes visible

    // S = Q K^T : per wave 16x64 = 4 (n) tiles, DK=64 in two K-chunks of 32
    v8f s[4];
#pragma unroll
    for (int n = 0; n < 4; ++n) {
      v8f acc = zero;
#pragma unroll
      for (int kc = 0; kc < 64; kc += 32) {
        v16bf a  = frag16x32(sQ, 64, 1, wq * 16, kc);
        v16bf bb = frag16x32(sK, 64, 1, n * 16, kc);   // B[k][n] = K[n][k]
        acc = wmma_bf16(a, bb, acc);
      }
      s[n] = acc;
    }

    // scale + relpos bias + causal/pad masks
#pragma unroll
    for (int n = 0; n < 4; ++n) {
      const int kcol = k0 + n * 16 + colh;
      const bool padk = spad[n * 16 + colh] != 0;
#pragma unroll
      for (int i = 0; i < 8; ++i) {
        const int qrow = q0 + wq * 16 + rbase + i;
        float x = s[n][i] * 0.125f + biash[(size_t)qrow * Tc + kcol];
        if (kcol > qrow || padk) x = -1.0e30f;
        s[n][i] = x;
      }
    }

    // row max across 64 cols: 4 tiles + 16-lane half-wave shuffle reduce
    float tm[8];
#pragma unroll
    for (int i = 0; i < 8; ++i)
      tm[i] = fmaxf(fmaxf(s[0][i], s[1][i]), fmaxf(s[2][i], s[3][i]));
#pragma unroll
    for (int off = 1; off < 16; off <<= 1)
#pragma unroll
      for (int i = 0; i < 8; ++i)
        tm[i] = fmaxf(tm[i], __shfl_xor(tm[i], off));

    float alpha[8], rs[8];
#pragma unroll
    for (int i = 0; i < 8; ++i) {
      float mn = fmaxf(mrow[i], tm[i]);
      alpha[i] = __expf(mrow[i] - mn);
      mrow[i] = mn;
      rs[i] = 0.0f;
    }
#pragma unroll
    for (int n = 0; n < 4; ++n)
#pragma unroll
      for (int i = 0; i < 8; ++i) {
        float p = __expf(s[n][i] - mrow[i]);
        s[n][i] = p;
        rs[i] += p;
      }
#pragma unroll
    for (int off = 1; off < 16; off <<= 1)
#pragma unroll
      for (int i = 0; i < 8; ++i)
        rs[i] += __shfl_xor(rs[i], off);
#pragma unroll
    for (int i = 0; i < 8; ++i) lrow[i] = lrow[i] * alpha[i] + rs[i];
#pragma unroll
    for (int j = 0; j < 4; ++j)
#pragma unroll
      for (int i = 0; i < 8; ++i) o[j][i] *= alpha[i];

    // P (bf16) -> per-wave LDS region, then O += P @ V
    __bf16* myP = sP[wq];
#pragma unroll
    for (int n = 0; n < 4; ++n)
#pragma unroll
      for (int i = 0; i < 8; ++i)
        myP[(rbase + i) * 64 + n * 16 + colh] = f2bf(s[n][i]);

#pragma unroll
    for (int j = 0; j < 4; ++j)
#pragma unroll
      for (int kc = 0; kc < 64; kc += 32) {
        v16bf a  = frag16x32(myP, 64, 1, 0, kc);
        v16bf bb = frag16x32(sV, 1, 64, j * 16, kc);   // B[k][n] = V[k][n]
        o[j] = wmma_bf16(a, bb, o[j]);
      }
  }

  // normalize and write merged-head output [B, T, D] (f32)
#pragma unroll
  for (int j = 0; j < 4; ++j)
#pragma unroll
    for (int i = 0; i < 8; ++i) {
      const int t = q0 + wq * 16 + rbase + i;
      const float inv = 1.0f / lrow[i];
      O[((size_t)b * Tc + t) * Dc + h * DKc + j * 16 + colh] = o[j][i] * inv;
    }
}

// ---------------------------------------------------------------------------
extern "C" void kernel_launch(void* const* d_in, const int* in_sizes, int n_in,
                              void* d_out, int out_size, void* d_ws, size_t ws_size,
                              hipStream_t stream) {
  (void)in_sizes; (void)n_in; (void)out_size; (void)ws_size;
  const float* q  = (const float*)d_in[0];
  const float* k  = (const float*)d_in[1];
  const float* v  = (const float*)d_in[2];
  const unsigned char* pad = (const unsigned char*)d_in[3];
  // d_in[4] = attn_mask (causal) — applied analytically in-kernel
  const float* bias = (const float*)d_in[5];
  const float* Wq = (const float*)d_in[6];
  const float* Wk = (const float*)d_in[7];
  const float* Wv = (const float*)d_in[8];
  const float* Wo = (const float*)d_in[9];
  float* out = (float*)d_out;

  const int M = Bc * Tc;                                    // 4096
  const size_t projElems = (size_t)Bc * Hc * Tc * DKc;      // 4M bf16 each
  unsigned short* Qp = (unsigned short*)d_ws;               //  8 MB
  unsigned short* Kp = Qp + projElems;                      //  8 MB
  unsigned short* Vp = Kp + projElems;                      //  8 MB
  float* attnO = (float*)(Vp + projElems);                  // 16 MB

  dim3 gg(Dc / 128, M / 128);
  gemm_xwt_wmma<<<gg, 256, 0, stream>>>(q, Wq, Qp, nullptr, M, Dc, Dc, 0);
  gemm_xwt_wmma<<<gg, 256, 0, stream>>>(k, Wk, Kp, nullptr, M, Dc, Dc, 0);
  gemm_xwt_wmma<<<gg, 256, 0, stream>>>(v, Wv, Vp, nullptr, M, Dc, Dc, 0);

  dim3 ga(Tc / 64, Hc, Bc);
  flash_attn_wmma<<<ga, 128, 0, stream>>>(Qp, Kp, Vp, bias, pad, attnO);

  gemm_xwt_wmma<<<gg, 256, 0, stream>>>(attnO, Wo, nullptr, out, M, Dc, Dc, 1);
}
